// BeliefPropagation_4509715661116
// MI455X (gfx1250) — compile-verified, matrix operand order checked by reading, not run on hardware
//
#include <hip/hip_runtime.h>
#include <math.h>

// Problem constants (from reference)
#define E_EDGES 6144
#define N_VARSC 2048
#define BATCH_B 128
#define N_ITERS 10
#define EPS_BP  1e-7f

// WMMA fragment / tile types
typedef _Float16 half8 __attribute__((ext_vector_type(8)));
typedef _Float16 v16h  __attribute__((ext_vector_type(16)));
typedef float    v8f   __attribute__((ext_vector_type(8)));
typedef unsigned int v4u __attribute__((ext_vector_type(4)));
typedef int          v8i __attribute__((ext_vector_type(8)));
typedef int          v4i __attribute__((ext_vector_type(4)));

// LDS f16 B-tile row stride in halves: 80B -> 16B aligned, conflict-free b128 reads
#define BSTRIDE 40

#if defined(__has_builtin)
#  if __has_builtin(__builtin_amdgcn_tensor_load_to_lds)
#    define HAVE_TDM 1
#  endif
#endif
#ifndef HAVE_TDM
#  define HAVE_TDM 0
#endif

// ---------------------------------------------------------------------------
// Small helper kernels
// ---------------------------------------------------------------------------
__global__ void cvt_f2h_kernel(const float* __restrict__ src,
                               _Float16* __restrict__ dst, int n) {
  int i = blockIdx.x * blockDim.x + threadIdx.x;
  if (i < n) dst[i] = (_Float16)src[i];
}

// col[e] = variable index of edge e (position of the 1 in llr_expander row e)
__global__ void compute_col_kernel(const float* __restrict__ expd,
                                   int* __restrict__ col) {
  int e = blockIdx.x * blockDim.x + threadIdx.x;
  if (e >= E_EDGES) return;
  const float* row = expd + (size_t)e * N_VARSC;
  int c = 0;
  for (int v = 0; v < N_VARSC; ++v) {
    if (row[v] != 0.0f) { c = v; break; }
  }
  col[e] = c;
}

// ve[v][0..2] = the 3 edges incident to variable v (deterministic order)
__global__ void build_ve_kernel(const float* __restrict__ mvf,
                                int* __restrict__ ve) {
  int v = blockIdx.x * blockDim.x + threadIdx.x;
  if (v >= N_VARSC) return;
  const float* row = mvf + (size_t)v * E_EDGES;
  int cnt = 0;
  int idx[3] = {0, 0, 0};
  for (int e = 0; e < E_EDGES; ++e) {
    if (row[e] != 0.0f) {
      if (cnt < 3) idx[cnt] = e;
      ++cnt;
    }
  }
  ve[v * 3 + 0] = idx[0];
  ve[v * 3 + 1] = idx[1];
  ve[v * 3 + 2] = idx[2];
}

// out[b,v] = 1 - sigmoid(sum_e x[b,e] - llr[b,v])
__global__ void finalize_kernel(const _Float16* __restrict__ Xh,
                                const int* __restrict__ ve,
                                const float* __restrict__ llr,
                                float* __restrict__ out) {
  int i = blockIdx.x * blockDim.x + threadIdx.x;
  if (i >= BATCH_B * N_VARSC) return;
  int b = i / N_VARSC;
  int v = i - b * N_VARSC;
  const _Float16* xr = Xh + (size_t)b * E_EDGES;
  float s = (float)xr[ve[v * 3 + 0]] + (float)xr[ve[v * 3 + 1]] +
            (float)xr[ve[v * 3 + 2]];
  float z = s - llr[i];
  out[i] = 1.0f / (1.0f + expf(z));
}

// ---------------------------------------------------------------------------
// Fragment loaders (documented CDNA5 16-bit WMMA layouts)
// ---------------------------------------------------------------------------
__device__ __forceinline__ v16h load_frag_global(const _Float16* p) {
  half8 lo = *(const half8*)(p);
  half8 hi = *(const half8*)(p + 16);
  v16h r;
#pragma unroll
  for (int i = 0; i < 8; ++i) { r[i] = lo[i]; r[i + 8] = hi[i]; }
  return r;
}

__device__ __forceinline__ v16h load_frag_lds(const _Float16* p) {
  half8 lo = *(const half8*)(p);
  half8 hi = *(const half8*)(p + 16);
  v16h r;
#pragma unroll
  for (int i = 0; i < 8; ++i) { r[i] = lo[i]; r[i + 8] = hi[i]; }
  return r;
}

// Force operands to be materialized in distinct registers NOW, so the four
// B-fragment ds_loads complete behind one wait and the WMMAs issue
// back-to-back (no per-WMMA s_wait_dscnt WAR stalls from tuple reuse).
__device__ __forceinline__ void pin_frags4(v16h& a, v16h& b, v16h& c, v16h& d) {
  asm volatile("" : "+v"(a), "+v"(b), "+v"(c), "+v"(d));
}

// Low 32 bits of a generic pointer into LDS = LDS byte address (ISA 10.2)
__device__ __forceinline__ unsigned lds_byte_addr(const void* p) {
  return (unsigned)(unsigned long long)p;
}

// Convert one staged f32 tile row-chunk (LDS) into the f16 B tile (LDS).
// 256 threads: thread t -> n = t>>2, ko = (t&3)*8 halves.
__device__ __forceinline__ void convert_tile_lds(const float* __restrict__ fsrc,
                                                 _Float16* __restrict__ Bs) {
  const int t  = threadIdx.x;
  const int nn = t >> 2;
  const int ko = (t & 3) << 3;
  const float* s = fsrc + nn * 32 + ko;
  float4 f0 = *(const float4*)(s);
  float4 f1 = *(const float4*)(s + 4);
  half8 h;
  h[0] = (_Float16)f0.x; h[1] = (_Float16)f0.y;
  h[2] = (_Float16)f0.z; h[3] = (_Float16)f0.w;
  h[4] = (_Float16)f1.x; h[5] = (_Float16)f1.y;
  h[6] = (_Float16)f1.z; h[7] = (_Float16)f1.w;
  *(half8*)(&Bs[nn * BSTRIDE + ko]) = h;
}

// Fallback synchronous staging: global f32 -> cvt -> LDS f16.
__device__ __forceinline__ void stage_mask_tile_sync(const float* __restrict__ mask,
                                                     int n0, int k,
                                                     _Float16* __restrict__ Bs) {
  const int t  = threadIdx.x;
  const int nn = t >> 2;
  const int ko = (t & 3) << 3;
  const float* src = mask + (size_t)(n0 + nn) * E_EDGES + k + ko;
  __builtin_prefetch(src + 32, 0, 0);
  float4 f0 = *(const float4*)(src);
  float4 f1 = *(const float4*)(src + 4);
  half8 h;
  h[0] = (_Float16)f0.x; h[1] = (_Float16)f0.y;
  h[2] = (_Float16)f0.z; h[3] = (_Float16)f0.w;
  h[4] = (_Float16)f1.x; h[5] = (_Float16)f1.y;
  h[6] = (_Float16)f1.z; h[7] = (_Float16)f1.w;
  *(half8*)(&Bs[nn * BSTRIDE + ko]) = h;
}

#if HAVE_TDM
// TDM: DMA one 64(n) x 32(k) f32 tile of the mask into LDS (packed [64][32]).
// D# group0: count=1 | lds_addr | global_addr | type=2.
// D# group1: data_size=4B, tensor_dim0/1 = E, tile_dim0=32, tile_dim1=64,
//            dim0_stride = E. Issued by one wave; tracked with TENSORcnt.
// clang-23 / therock-10.0 6-arg builtin:
//   (uint32x4 g0, int32x8 g1, int32x4 g2, int32x4 g3, int32x8 g4, i32 cpol)
__device__ __forceinline__ void tdm_issue_tile(const float* __restrict__ mask,
                                               int n0, int k,
                                               unsigned lds_off) {
  unsigned long long ga =
      (unsigned long long)(const void*)(mask + (size_t)n0 * E_EDGES + k);
  v4u g0 = { 1u,                                   // count=1 (valid descriptor)
             lds_off,                              // lds_addr
             (unsigned)ga,                         // global_addr[31:0]
             ((unsigned)(ga >> 32) & 0x01FFFFFFu)  // global_addr[56:32]
                 | 0x80000000u };                  // type=2 at bits [127:126]
  v8i g1 = { (int)(2u << 16),          // data_size=2 (4 bytes); wg_mask=0
             (int)(6144u << 16),       // tensor_dim0[15:0] at bits [63:48]
             (int)(6144u << 16),       // tensor_dim1[15:0] at bits [95:80]
             (int)(32u << 16),         // tile_dim0 = 32 at bits [127:112]
             64,                       // tile_dim1 = 64 at bits [143:128]
             6144,                     // tensor_dim0_stride[31:0]
             0, 0 };
  v4i gz4 = { 0, 0, 0, 0 };
  v8i gz8 = { 0, 0, 0, 0, 0, 0, 0, 0 };
  __builtin_amdgcn_tensor_load_to_lds(g0, g1, gz4, gz4, gz8, 0);
}
#endif

// Per-lane async copy: 512 b128 units per 64x32 f32 tile; 2 units per thread.
__device__ __forceinline__ void async_tile_issue(const float* __restrict__ mask,
                                                 int n0, int k,
                                                 float* __restrict__ fdst) {
  const int t = threadIdx.x;
#pragma unroll
  for (int u = 0; u < 2; ++u) {
    const int unit = t + (u << 8);    // 0..511
    const int nn = unit >> 3;
    const int ko = (unit & 7) << 2;   // float offset within row
    const float* g = mask + (size_t)(n0 + nn) * E_EDGES + k + ko;
    unsigned loff = lds_byte_addr(fdst + nn * 32 + ko);
    asm volatile("global_load_async_to_lds_b128 %0, %1, off"
                 :: "v"(loff), "v"(g)
                 : "memory");
  }
}

__device__ __forceinline__ void async_wait_all() {
  asm volatile("s_wait_asynccnt 0x0" ::: "memory");
}

// ---------------------------------------------------------------------------
// GEMM 1 (variable->check): t = tanh(X @ mask_v - llr_e) -> logmag/parity.
// B-tile staged by the Tensor Data Mover (double-buffered f32 in LDS).
// Workgroup: 8 waves, tile M=128 x N=64. Grid = E/64 = 96.
// ---------------------------------------------------------------------------
__global__ __launch_bounds__(256)
void bp_vc_wmma_kernel(const _Float16* __restrict__ Xh,
                       const float* __restrict__ maskv,
                       const int* __restrict__ col,
                       const float* __restrict__ llr,
                       _Float16* __restrict__ logmag,
                       _Float16* __restrict__ parity) {
  __shared__ __align__(16) float    Fbuf[2][64 * 32];
  __shared__ __align__(16) _Float16 Bs[64 * BSTRIDE];
  const int n0   = blockIdx.x * 64;
  const int wave = threadIdx.x >> 5;
  const int lane = threadIdx.x & 31;
  const int m0   = wave * 16;
  const int rowi = lane & 15;
  const int ksel = (lane >> 4) << 3;  // 0 or 8

  v8f acc[4] = {};

#if HAVE_TDM
  if (threadIdx.x < 32)
    tdm_issue_tile(maskv, n0, 0, lds_byte_addr(&Fbuf[0][0]));
#endif

  int buf = 0;
  for (int k = 0; k < E_EDGES; k += 32, buf ^= 1) {
#if HAVE_TDM
    if (threadIdx.x < 32) __builtin_amdgcn_s_wait_tensorcnt(0);
    __syncthreads();  // tile k resident in Fbuf[buf]; prev Bs readers done
    if ((k + 32) < E_EDGES && threadIdx.x < 32)
      tdm_issue_tile(maskv, n0, k + 32, lds_byte_addr(&Fbuf[buf ^ 1][0]));
    convert_tile_lds(&Fbuf[buf][0], Bs);
#else
    stage_mask_tile_sync(maskv, n0, k, Bs);
#endif
    __syncthreads();

    v16h a = load_frag_global(Xh + (size_t)(m0 + rowi) * E_EDGES + k + ksel);

    v16h b0 = load_frag_lds(&Bs[(0 * 16 + rowi) * BSTRIDE + ksel]);
    v16h b1 = load_frag_lds(&Bs[(1 * 16 + rowi) * BSTRIDE + ksel]);
    v16h b2 = load_frag_lds(&Bs[(2 * 16 + rowi) * BSTRIDE + ksel]);
    v16h b3 = load_frag_lds(&Bs[(3 * 16 + rowi) * BSTRIDE + ksel]);
    pin_frags4(b0, b1, b2, b3);  // distinct tuples -> back-to-back WMMAs
    acc[0] = __builtin_amdgcn_wmma_f32_16x16x32_f16(false, a, false, b0,
                                                    (short)0, acc[0], false, false);
    acc[1] = __builtin_amdgcn_wmma_f32_16x16x32_f16(false, a, false, b1,
                                                    (short)0, acc[1], false, false);
    acc[2] = __builtin_amdgcn_wmma_f32_16x16x32_f16(false, a, false, b2,
                                                    (short)0, acc[2], false, false);
    acc[3] = __builtin_amdgcn_wmma_f32_16x16x32_f16(false, a, false, b3,
                                                    (short)0, acc[3], false, false);
    __syncthreads();
  }

  const int rbase = (lane >> 4) << 3;
#pragma unroll
  for (int t = 0; t < 4; ++t) {
    const int ng = n0 + t * 16 + rowi;
    const int cv = col[ng];
#pragma unroll
    for (int j = 0; j < 8; ++j) {
      const int bb = m0 + rbase + j;
      float z  = acc[t][j] - llr[bb * N_VARSC + cv];
      float th = tanhf(z);
      logmag[(size_t)bb * E_EDGES + ng] = (_Float16)logf(fabsf(th) + EPS_BP);
      parity[(size_t)bb * E_EDGES + ng] = (_Float16)(th < 0.0f ? 1.0f : 0.0f);
    }
  }
}

// ---------------------------------------------------------------------------
// GEMM 2 (check->variable): dual accumulators share every B fragment.
// B-tile staged with per-lane GLOBAL_LOAD_ASYNC_TO_LDS_B128 (ASYNCcnt).
// ---------------------------------------------------------------------------
__global__ __launch_bounds__(256)
void bp_cv_wmma_kernel(const _Float16* __restrict__ logmag,
                       const _Float16* __restrict__ parity,
                       const float* __restrict__ maskc,
                       const int* __restrict__ clampp,
                       _Float16* __restrict__ Xh) {
  __shared__ __align__(16) float    Fbuf[2][64 * 32];
  __shared__ __align__(16) _Float16 Bs[64 * BSTRIDE];
  const int n0   = blockIdx.x * 64;
  const int wave = threadIdx.x >> 5;
  const int lane = threadIdx.x & 31;
  const int m0   = wave * 16;
  const int rowi = lane & 15;
  const int ksel = (lane >> 4) << 3;

  const float cmax = (float)(*clampp);

  v8f accL[4] = {};
  v8f accP[4] = {};

  async_tile_issue(maskc, n0, 0, &Fbuf[0][0]);

  int buf = 0;
  for (int k = 0; k < E_EDGES; k += 32, buf ^= 1) {
    async_wait_all();   // this wave's async copies for tile k complete
    __syncthreads();    // all waves' portions resident; prev Bs readers done
    if ((k + 32) < E_EDGES)
      async_tile_issue(maskc, n0, k + 32, &Fbuf[buf ^ 1][0]);
    convert_tile_lds(&Fbuf[buf][0], Bs);
    __syncthreads();

    const size_t aoff = (size_t)(m0 + rowi) * E_EDGES + k + ksel;
    v16h aL = load_frag_global(logmag + aoff);
    v16h aP = load_frag_global(parity + aoff);

    v16h b0 = load_frag_lds(&Bs[(0 * 16 + rowi) * BSTRIDE + ksel]);
    v16h b1 = load_frag_lds(&Bs[(1 * 16 + rowi) * BSTRIDE + ksel]);
    v16h b2 = load_frag_lds(&Bs[(2 * 16 + rowi) * BSTRIDE + ksel]);
    v16h b3 = load_frag_lds(&Bs[(3 * 16 + rowi) * BSTRIDE + ksel]);
    pin_frags4(b0, b1, b2, b3);  // distinct tuples -> unstalled WMMA chain
    accL[0] = __builtin_amdgcn_wmma_f32_16x16x32_f16(false, aL, false, b0,
                                                     (short)0, accL[0], false, false);
    accP[0] = __builtin_amdgcn_wmma_f32_16x16x32_f16(false, aP, false, b0,
                                                     (short)0, accP[0], false, false);
    accL[1] = __builtin_amdgcn_wmma_f32_16x16x32_f16(false, aL, false, b1,
                                                     (short)0, accL[1], false, false);
    accP[1] = __builtin_amdgcn_wmma_f32_16x16x32_f16(false, aP, false, b1,
                                                     (short)0, accP[1], false, false);
    accL[2] = __builtin_amdgcn_wmma_f32_16x16x32_f16(false, aL, false, b2,
                                                     (short)0, accL[2], false, false);
    accP[2] = __builtin_amdgcn_wmma_f32_16x16x32_f16(false, aP, false, b2,
                                                     (short)0, accP[2], false, false);
    accL[3] = __builtin_amdgcn_wmma_f32_16x16x32_f16(false, aL, false, b3,
                                                     (short)0, accL[3], false, false);
    accP[3] = __builtin_amdgcn_wmma_f32_16x16x32_f16(false, aP, false, b3,
                                                     (short)0, accP[3], false, false);
    __syncthreads();
  }

  const int rbase = (lane >> 4) << 3;
#pragma unroll
  for (int t = 0; t < 4; ++t) {
    const int ng = n0 + t * 16 + rowi;
#pragma unroll
    for (int j = 0; j < 8; ++j) {
      const int bb   = m0 + rbase + j;
      float s1   = accL[t][j];
      float s2   = accP[t][j];                    // exact small integer
      float sign = 1.0f - 2.0f * fmodf(s2, 2.0f);
      float prod = sign * expf(s1);
      prod = fminf(fmaxf(prod, -1.0f + EPS_BP), 1.0f - EPS_BP);
      float cv2  = log1pf(prod) - log1pf(-prod);  // 2*atanh
      float xv   = fminf(fmaxf(cv2, -cmax), cmax);
      Xh[(size_t)bb * E_EDGES + ng] = (_Float16)xv;
    }
  }
}

// ---------------------------------------------------------------------------
// Host-side launch sequence (all on `stream`, graph-capture safe)
// ---------------------------------------------------------------------------
extern "C" void kernel_launch(void* const* d_in, const int* in_sizes, int n_in,
                              void* d_out, int out_size, void* d_ws,
                              size_t ws_size, hipStream_t stream) {
  (void)in_sizes; (void)n_in; (void)out_size; (void)ws_size;

  const float* x0      = (const float*)d_in[0];
  const float* llr     = (const float*)d_in[1];
  const float* mask_v  = (const float*)d_in[2];
  const float* mask_c  = (const float*)d_in[3];
  const float* llr_exp = (const float*)d_in[4];
  const float* mvf     = (const float*)d_in[5];
  const int*   clampp  = (const int*)d_in[6];
  float* out = (float*)d_out;

  char* ws = (char*)d_ws;
  _Float16* Xh     = (_Float16*)(ws);
  _Float16* logmag = (_Float16*)(ws + 1572864);
  _Float16* parity = (_Float16*)(ws + 2 * 1572864);
  int*      col    = (int*)(ws + 3 * 1572864);
  int*      ve     = (int*)(ws + 3 * 1572864 + 24576);

  const int BE = BATCH_B * E_EDGES;
  const int BN = BATCH_B * N_VARSC;

  compute_col_kernel<<<(E_EDGES + 255) / 256, 256, 0, stream>>>(llr_exp, col);
  build_ve_kernel<<<(N_VARSC + 255) / 256, 256, 0, stream>>>(mvf, ve);
  cvt_f2h_kernel<<<(BE + 255) / 256, 256, 0, stream>>>(x0, Xh, BE);

  for (int it = 0; it < N_ITERS; ++it) {
    bp_vc_wmma_kernel<<<E_EDGES / 64, 256, 0, stream>>>(
        Xh, mask_v, col, llr, logmag, parity);
    bp_cv_wmma_kernel<<<E_EDGES / 64, 256, 0, stream>>>(
        logmag, parity, mask_c, clampp, Xh);
  }

  finalize_kernel<<<(BN + 255) / 256, 256, 0, stream>>>(Xh, ve, llr, out);
}
